// CustomLSTM2_78486232367300
// MI455X (gfx1250) — matrix operand order repaired and checked by reading, not compiled, not run on hardware
//
#include <hip/hip_runtime.h>

#define T_STEPS 512

typedef __attribute__((ext_vector_type(16))) __bf16 v16bf;
typedef __attribute__((ext_vector_type(8)))  float  v8f;

union FragU { v16bf v; unsigned short u[16]; uint4 q[2]; };

// ---- workspace layout (in bf16 halves) ----
#define O_ET1 0           // 256x256
#define O_W1  65536       // 4 x 256x544
#define W1SZ  139264
#define O_ET2 622592      // 256x256
#define O_W2  688128      // 4 x 256x768
#define W2SZ  196608
#define WS_BYTES_BF 2949120

__device__ __forceinline__ unsigned short f2bf(float f){
  unsigned int u = __float_as_uint(f);
  u += 0x7FFFu + ((u >> 16) & 1u);
  return (unsigned short)(u >> 16);
}

__device__ __forceinline__ float fexp2(float x){
#if __has_builtin(__builtin_amdgcn_exp2f)
  return __builtin_amdgcn_exp2f(x);      // v_exp_f32, branch-free
#else
  return exp2f(x);
#endif
}
__device__ __forceinline__ float frcp(float x){
#if __has_builtin(__builtin_amdgcn_rcpf)
  return __builtin_amdgcn_rcpf(x);       // v_rcp_f32, branch-free
#else
  return 1.f / x;
#endif
}
__device__ __forceinline__ float fast_tanh(float x){
#if __has_builtin(__builtin_amdgcn_tanhf)
  return __builtin_amdgcn_tanhf(x);      // v_tanh_f32 if toolchain exposes it
#elif __has_builtin(__builtin_amdgcn_tanh_f32)
  return __builtin_amdgcn_tanh_f32(x);
#else
  // tanh(x) = sign(x) * (1 - e) / (1 + e), e = exp(-2|x|) = 2^(-2|x|*log2(e))
  float ax = __builtin_fabsf(x);
  float e  = fexp2(ax * -2.885390081777927f);
  float r  = (1.f - e) * frcp(1.f + e);
  return __builtin_copysignf(r, x);
#endif
}
__device__ __forceinline__ float sigm(float x){
  // 1/(1+exp(-x)) = rcp(1 + 2^(-x*log2(e)))
  float e = fexp2(x * -1.4426950408889634f);
  return frcp(1.f + e);
}

__device__ __forceinline__ v8f wmma_bf16(v16bf a, v16bf b, v8f c){
  return __builtin_amdgcn_wmma_f32_16x16x32_bf16(false, a, false, b, (short)0, c, false, false);
}

// A fragment (16x32 bf16) from an LDS row-major tile, rows=16, row stride = `stride` halves.
// Layout: lane<16 -> K 0..15; lane>=16 -> K 16..31 (contiguous, matches load_b_gl).
__device__ __forceinline__ v16bf load_a_lds(const unsigned short* base, int stride, int kc, int lane){
  FragU a;
  int m = lane & 15, hi = (lane >> 4) & 1;
  #pragma unroll
  for (int j = 0; j < 8; ++j){
    int kb = kc + (hi ? 16 : 0) + (j << 1);
    const unsigned short* p = base + m * stride + kb;
    a.u[2*j]   = p[0];
    a.u[2*j+1] = p[1];
  }
  return a.v;
}

// B fragment (32x16 bf16) for tile (nt,kt) from bf16 weight matrix stored row-major (N=256 rows, Kp cols).
// Per lane: column n = nt*16 + lane%16, contiguous K run of 16 starting at kt*32 + (lane/16)*16.
__device__ __forceinline__ v16bf load_b_gl(const unsigned short* wb, int Kp, int nt, int kt, int lane){
  const unsigned short* p = wb + (size_t)(nt*16 + (lane & 15)) * Kp + kt*32 + ((lane >> 4) << 4);
  FragU b;
  b.q[0] = ((const uint4*)p)[0];
  b.q[1] = ((const uint4*)p)[1];
  return b.v;
}

// ---------- prep kernels ----------
__global__ void prep_convert(const float* __restrict__ src, unsigned short* __restrict__ dst,
                             int N, int Kreal, int Kp){
  int total = N * Kp;
  for (int idx = blockIdx.x*blockDim.x + threadIdx.x; idx < total; idx += gridDim.x*blockDim.x){
    int n = idx / Kp, k = idx - n*Kp;
    float v = (k < Kreal) ? src[(size_t)n*Kreal + k] : 0.f;
    dst[idx] = f2bf(v);
  }
}
__global__ void prep_zero(float* __restrict__ p, int n){
  for (int i = blockIdx.x*blockDim.x + threadIdx.x; i < n; i += gridDim.x*blockDim.x) p[i] = 0.f;
}

// ---------- main persistent kernel: 16 blocks x 256 threads, each block owns 16 batch rows ----------
__global__ __launch_bounds__(256, 1) void lstm2_main(
    const float* __restrict__ x,
    const unsigned short* __restrict__ wsb,
    float* __restrict__ he_g, float* __restrict__ he2_g,
    const float* __restrict__ bi1, const float* __restrict__ bf1,
    const float* __restrict__ bc1, const float* __restrict__ bo1,
    const float* __restrict__ ln1w, const float* __restrict__ ln1b,
    const float* __restrict__ bi2, const float* __restrict__ bf2,
    const float* __restrict__ bc2, const float* __restrict__ bo2,
    const float* __restrict__ ln2w, const float* __restrict__ ln2b,
    const float* __restrict__ fcw,
    float* __restrict__ out)
{
  __shared__ float sh_h [16][256];
  __shared__ float sh_c [16][256];
  __shared__ float sh_h2[16][256];
  __shared__ float sh_c2[16][256];
  __shared__ float sh_u [16][256];
  __shared__ unsigned short sh_hc1[16][544];   // [xt(16) | te(256) | hn(256) | pad(16)]
  __shared__ unsigned short sh_hc2[16][768];   // [h(256) | te2(256) | hn2(256)]
  __shared__ unsigned short sh_heb[16][256];   // bf16 staging for he / he2
  __shared__ float sh_ln1w[256], sh_ln1b[256], sh_ln2w[256], sh_ln2b[256];
  __shared__ float sh_b1[4][256], sh_b2[4][256];
  __shared__ float sh_fc[257];
  __shared__ float sh_dd[16], sh_px[16], sh_cx[16], sh_prev[16], sh_en[16];

  const int tid  = threadIdx.x;
  const int lane = tid & 31;
  const int wv   = tid >> 5;
  const int b0   = blockIdx.x * 16;

  { // constants + state init
    int i = tid;
    sh_ln1w[i] = ln1w[i]; sh_ln1b[i] = ln1b[i]; sh_ln2w[i] = ln2w[i]; sh_ln2b[i] = ln2b[i];
    sh_b1[0][i] = bi1[i]; sh_b1[1][i] = bf1[i]; sh_b1[2][i] = bc1[i]; sh_b1[3][i] = bo1[i];
    sh_b2[0][i] = bi2[i]; sh_b2[1][i] = bf2[i]; sh_b2[2][i] = bc2[i]; sh_b2[3][i] = bo2[i];
    for (int j = tid; j < 257; j += 256) sh_fc[j] = fcw[j];
    for (int j = tid; j < 16*256; j += 256){
      int r = j >> 8, n = j & 255;
      sh_h[r][n] = 0.f; sh_c[r][n] = 0.f; sh_h2[r][n] = 0.f; sh_c2[r][n] = 0.f;
    }
    for (int j = tid; j < 16*16; j += 256){ sh_hc1[j >> 4][528 + (j & 15)] = 0; }
    if (tid < 16){ sh_px[tid] = 0.f; sh_prev[tid] = 0.f; sh_en[tid] = 0.f; }
  }
  __syncthreads();

  #pragma unroll 1
  for (int t = 0; t < T_STEPS; ++t){
    // ---------------- P1: dd, xt, LN(h)->hc1, bf16(he) ----------------
    {
      int r = tid >> 4, j0 = tid & 15;
      const float* xrow = x + ((size_t)(b0 + r) * T_STEPS + t) * 16;
      if (j0 == 0){ float cx = xrow[0]; sh_cx[r] = cx; sh_dd[r] = cx - sh_px[r]; }
      sh_hc1[r][j0] = f2bf(xrow[j0]);
      float s = 0.f, ss = 0.f;
      #pragma unroll
      for (int k2 = 0; k2 < 16; ++k2){ float v = sh_h[r][j0 + 16*k2]; s += v; ss += v*v; }
      #pragma unroll
      for (int m = 8; m; m >>= 1){ s += __shfl_xor(s, m, 32); ss += __shfl_xor(ss, m, 32); }
      float mu = s * (1.f/256.f), var = ss * (1.f/256.f) - mu*mu, rs = rsqrtf(var + 1e-5f);
      #pragma unroll
      for (int k2 = 0; k2 < 16; ++k2){
        int n = j0 + 16*k2;
        sh_hc1[r][272 + n] = f2bf((sh_h[r][n] - mu) * rs * sh_ln1w[n] + sh_ln1b[n]);
      }
      const float* hesrc = he_g + (size_t)(b0 + r) * 256;
      #pragma unroll
      for (int k2 = 0; k2 < 16; ++k2){ int n = j0 + 16*k2; sh_heb[r][n] = f2bf(hesrc[n]); }
    }
    __syncthreads();
    // ---------------- P2: te1 = tanh(he @ et1^T) -> hc1 (double-buffered) ----------------
    #pragma unroll
    for (int q = 0; q < 2; ++q){
      int nt = wv*2 + q;
      v8f acc = {0.f,0.f,0.f,0.f,0.f,0.f,0.f,0.f};
      v16bf a = load_a_lds(&sh_heb[0][0], 256, 0, lane);
      v16bf b = load_b_gl(wsb + O_ET1, 256, nt, 0, lane);
      #pragma unroll 2
      for (int kt = 0; kt < 8; ++kt){
        int ktn = (kt + 1 < 8) ? kt + 1 : kt;
        v16bf an = load_a_lds(&sh_heb[0][0], 256, ktn*32, lane);
        v16bf bn = load_b_gl(wsb + O_ET1, 256, nt, ktn, lane);
        acc = wmma_bf16(a, b, acc);
        a = an; b = bn;
      }
      int n = nt*16 + (lane & 15), mb = (lane >> 4) * 8;
      #pragma unroll
      for (int e = 0; e < 8; ++e) sh_hc1[mb + e][16 + n] = f2bf(fast_tanh(acc[e]));
    }
    __syncthreads();
    // ---------------- P3: layer-1 gates, c/u update (double-buffered) ----------------
    #pragma unroll
    for (int q = 0; q < 2; ++q){
      int nt = wv*2 + q;
      v8f ai = {0.f,0.f,0.f,0.f,0.f,0.f,0.f,0.f};
      v8f af = ai, ag = ai, ao = ai;
      v16bf a  = load_a_lds(&sh_hc1[0][0], 544, 0, lane);
      v16bf b0 = load_b_gl(wsb + O_W1 + 0*W1SZ, 544, nt, 0, lane);
      v16bf b1 = load_b_gl(wsb + O_W1 + 1*W1SZ, 544, nt, 0, lane);
      v16bf b2 = load_b_gl(wsb + O_W1 + 2*W1SZ, 544, nt, 0, lane);
      v16bf b3 = load_b_gl(wsb + O_W1 + 3*W1SZ, 544, nt, 0, lane);
      #pragma unroll 2
      for (int kt = 0; kt < 17; ++kt){
        int ktn = (kt + 1 < 17) ? kt + 1 : kt;
        v16bf an = load_a_lds(&sh_hc1[0][0], 544, ktn*32, lane);
        v16bf c0 = load_b_gl(wsb + O_W1 + 0*W1SZ, 544, nt, ktn, lane);
        v16bf c1 = load_b_gl(wsb + O_W1 + 1*W1SZ, 544, nt, ktn, lane);
        v16bf c2 = load_b_gl(wsb + O_W1 + 2*W1SZ, 544, nt, ktn, lane);
        v16bf c3 = load_b_gl(wsb + O_W1 + 3*W1SZ, 544, nt, ktn, lane);
        ai = wmma_bf16(a, b0, ai);
        af = wmma_bf16(a, b1, af);
        ag = wmma_bf16(a, b2, ag);
        ao = wmma_bf16(a, b3, ao);
        a = an; b0 = c0; b1 = c1; b2 = c2; b3 = c3;
      }
      int n = nt*16 + (lane & 15), mb = (lane >> 4) * 8;
      #pragma unroll
      for (int e = 0; e < 8; ++e){
        int m = mb + e;
        float iv = sigm(ai[e] + sh_b1[0][n]);
        float fv = sigm(af[e] + sh_b1[1][n]);
        float gv = fast_tanh(ag[e] + sh_b1[2][n]);
        float ov = sigm(ao[e] + sh_b1[3][n]);
        float cn = fv * sh_c[m][n] + iv * gv;
        sh_c[m][n] = cn;
        sh_u[m][n] = ov * fast_tanh(cn) + sh_h[m][n];
      }
    }
    __syncthreads();
    // ---------------- P4: LN(u)->h (+he), LN(h2)->hc2, bf16(he2) ----------------
    {
      int r = tid >> 4, j0 = tid & 15;
      float s = 0.f, ss = 0.f;
      #pragma unroll
      for (int k2 = 0; k2 < 16; ++k2){ float v = sh_u[r][j0 + 16*k2]; s += v; ss += v*v; }
      #pragma unroll
      for (int m = 8; m; m >>= 1){ s += __shfl_xor(s, m, 32); ss += __shfl_xor(ss, m, 32); }
      float mu = s * (1.f/256.f), var = ss * (1.f/256.f) - mu*mu, rs = rsqrtf(var + 1e-5f);
      float ddv = sh_dd[r];
      float* hedst = he_g + (size_t)(b0 + r) * 256;
      #pragma unroll
      for (int k2 = 0; k2 < 16; ++k2){
        int n = j0 + 16*k2;
        float hn   = (sh_u[r][n] - mu) * rs * sh_ln1w[n] + sh_ln1b[n];
        float hold = sh_h[r][n];
        sh_h[r][n] = hn;
        hedst[n] += (hn + hold) * 0.5f * ddv;
        sh_hc2[r][n] = f2bf(hn);
      }
      s = 0.f; ss = 0.f;
      #pragma unroll
      for (int k2 = 0; k2 < 16; ++k2){ float v = sh_h2[r][j0 + 16*k2]; s += v; ss += v*v; }
      #pragma unroll
      for (int m = 8; m; m >>= 1){ s += __shfl_xor(s, m, 32); ss += __shfl_xor(ss, m, 32); }
      float mu2 = s * (1.f/256.f), var2 = ss * (1.f/256.f) - mu2*mu2, rs2 = rsqrtf(var2 + 1e-5f);
      #pragma unroll
      for (int k2 = 0; k2 < 16; ++k2){
        int n = j0 + 16*k2;
        sh_hc2[r][512 + n] = f2bf((sh_h2[r][n] - mu2) * rs2 * sh_ln2w[n] + sh_ln2b[n]);
      }
      const float* he2src = he2_g + (size_t)(b0 + r) * 256;
      #pragma unroll
      for (int k2 = 0; k2 < 16; ++k2){ int n = j0 + 16*k2; sh_heb[r][n] = f2bf(he2src[n]); }
    }
    __syncthreads();
    // ---------------- P5: te2 -> hc2 (double-buffered) ----------------
    #pragma unroll
    for (int q = 0; q < 2; ++q){
      int nt = wv*2 + q;
      v8f acc = {0.f,0.f,0.f,0.f,0.f,0.f,0.f,0.f};
      v16bf a = load_a_lds(&sh_heb[0][0], 256, 0, lane);
      v16bf b = load_b_gl(wsb + O_ET2, 256, nt, 0, lane);
      #pragma unroll 2
      for (int kt = 0; kt < 8; ++kt){
        int ktn = (kt + 1 < 8) ? kt + 1 : kt;
        v16bf an = load_a_lds(&sh_heb[0][0], 256, ktn*32, lane);
        v16bf bn = load_b_gl(wsb + O_ET2, 256, nt, ktn, lane);
        acc = wmma_bf16(a, b, acc);
        a = an; b = bn;
      }
      int n = nt*16 + (lane & 15), mb = (lane >> 4) * 8;
      #pragma unroll
      for (int e = 0; e < 8; ++e) sh_hc2[mb + e][256 + n] = f2bf(fast_tanh(acc[e]));
    }
    __syncthreads();
    // ---------------- P6: layer-2 gates, c2/u update (double-buffered) ----------------
    #pragma unroll
    for (int q = 0; q < 2; ++q){
      int nt = wv*2 + q;
      v8f ai = {0.f,0.f,0.f,0.f,0.f,0.f,0.f,0.f};
      v8f af = ai, ag = ai, ao = ai;
      v16bf a  = load_a_lds(&sh_hc2[0][0], 768, 0, lane);
      v16bf b0 = load_b_gl(wsb + O_W2 + 0*W2SZ, 768, nt, 0, lane);
      v16bf b1 = load_b_gl(wsb + O_W2 + 1*W2SZ, 768, nt, 0, lane);
      v16bf b2 = load_b_gl(wsb + O_W2 + 2*W2SZ, 768, nt, 0, lane);
      v16bf b3 = load_b_gl(wsb + O_W2 + 3*W2SZ, 768, nt, 0, lane);
      #pragma unroll 2
      for (int kt = 0; kt < 24; ++kt){
        int ktn = (kt + 1 < 24) ? kt + 1 : kt;
        v16bf an = load_a_lds(&sh_hc2[0][0], 768, ktn*32, lane);
        v16bf c0 = load_b_gl(wsb + O_W2 + 0*W2SZ, 768, nt, ktn, lane);
        v16bf c1 = load_b_gl(wsb + O_W2 + 1*W2SZ, 768, nt, ktn, lane);
        v16bf c2 = load_b_gl(wsb + O_W2 + 2*W2SZ, 768, nt, ktn, lane);
        v16bf c3 = load_b_gl(wsb + O_W2 + 3*W2SZ, 768, nt, ktn, lane);
        ai = wmma_bf16(a, b0, ai);
        af = wmma_bf16(a, b1, af);
        ag = wmma_bf16(a, b2, ag);
        ao = wmma_bf16(a, b3, ao);
        a = an; b0 = c0; b1 = c1; b2 = c2; b3 = c3;
      }
      int n = nt*16 + (lane & 15), mb = (lane >> 4) * 8;
      #pragma unroll
      for (int e = 0; e < 8; ++e){
        int m = mb + e;
        float iv = sigm(ai[e] + sh_b2[0][n]);
        float fv = sigm(af[e] + sh_b2[1][n]);
        float gv = fast_tanh(ag[e] + sh_b2[2][n]);
        float ov = sigm(ao[e] + sh_b2[3][n]);
        float cn = fv * sh_c2[m][n] + iv * gv;
        sh_c2[m][n] = cn;
        sh_u[m][n] = ov * fast_tanh(cn) + sh_h2[m][n];
      }
    }
    __syncthreads();
    // ---------------- P7: LN(u)->h2 (+he2) ----------------
    {
      int r = tid >> 4, j0 = tid & 15;
      float s = 0.f, ss = 0.f;
      #pragma unroll
      for (int k2 = 0; k2 < 16; ++k2){ float v = sh_u[r][j0 + 16*k2]; s += v; ss += v*v; }
      #pragma unroll
      for (int m = 8; m; m >>= 1){ s += __shfl_xor(s, m, 32); ss += __shfl_xor(ss, m, 32); }
      float mu = s * (1.f/256.f), var = ss * (1.f/256.f) - mu*mu, rs = rsqrtf(var + 1e-5f);
      float ddv = sh_dd[r];
      float* he2dst = he2_g + (size_t)(b0 + r) * 256;
      #pragma unroll
      for (int k2 = 0; k2 < 16; ++k2){
        int n = j0 + 16*k2;
        float hn   = (sh_u[r][n] - mu) * rs * sh_ln2w[n] + sh_ln2b[n];
        float hold = sh_h2[r][n];
        sh_h2[r][n] = hn;
        he2dst[n] += (hn + hold) * 0.5f * ddv;
      }
    }
    __syncthreads();
    // ---------------- P8: fc output + energy integrator, emit (parallel dot) ----------------
    {
      int r = tid >> 4, j0 = tid & 15;
      float part = 0.f;
      #pragma unroll
      for (int k2 = 0; k2 < 16; ++k2){ int n = j0 + 16*k2; part += sh_h2[r][n] * sh_fc[n]; }
      #pragma unroll
      for (int m = 8; m; m >>= 1) part += __shfl_xor(part, m, 32);
      if (j0 == 0){
        float acc = part + sh_fc[256] * sh_cx[r];
        float env = sh_en[r] + (acc + sh_prev[r]) * 0.5f * sh_dd[r];
        sh_en[r] = env; sh_prev[r] = acc; sh_px[r] = sh_cx[r];
        out[(size_t)(b0 + r) * T_STEPS + t]          = acc;
        out[131072 + (size_t)(b0 + r) * T_STEPS + t] = env;
      }
    }
    __syncthreads();
  }

  // ---------------- carry ----------------
  {
    const size_t C0 = 262144; // 2*B*T
    for (int j = tid; j < 4096; j += 256){
      int r = j >> 8, n = j & 255;
      size_t g = (size_t)(b0 + r) * 256 + n;
      out[C0 +      0 + g] = sh_h [r][n];
      out[C0 +  65536 + g] = sh_h2[r][n];
      out[C0 + 131072 + g] = sh_c [r][n];
      out[C0 + 196608 + g] = sh_c2[r][n];
      out[C0 + 262400 + g] = he_g [g];
      out[C0 + 327936 + g] = he2_g[g];
    }
    if (tid < 16){
      size_t g = (size_t)b0 + tid;
      out[C0 + 262144 + g] = sh_prev[tid];
      out[C0 + 393472 + g] = sh_en[tid];
      out[C0 + 393728 + g] = sh_cx[tid];
    }
  }
}

extern "C" void kernel_launch(void* const* d_in, const int* in_sizes, int n_in,
                              void* d_out, int out_size, void* d_ws, size_t ws_size,
                              hipStream_t stream) {
  (void)in_sizes; (void)n_in; (void)out_size; (void)ws_size;
  const float* x     = (const float*)d_in[0];
  const float* et1_w = (const float*)d_in[1];
  const float* wi1   = (const float*)d_in[2];
  const float* bi1   = (const float*)d_in[3];
  const float* wf1   = (const float*)d_in[4];
  const float* bf1   = (const float*)d_in[5];
  const float* wc1   = (const float*)d_in[6];
  const float* bc1   = (const float*)d_in[7];
  const float* wo1   = (const float*)d_in[8];
  const float* bo1   = (const float*)d_in[9];
  const float* ln1w  = (const float*)d_in[10];
  const float* ln1b  = (const float*)d_in[11];
  const float* et2_w = (const float*)d_in[12];
  const float* wi2   = (const float*)d_in[13];
  const float* bi2   = (const float*)d_in[14];
  const float* wf2   = (const float*)d_in[15];
  const float* bf2   = (const float*)d_in[16];
  const float* wc2   = (const float*)d_in[17];
  const float* bc2   = (const float*)d_in[18];
  const float* wo2   = (const float*)d_in[19];
  const float* bo2   = (const float*)d_in[20];
  const float* ln2w  = (const float*)d_in[21];
  const float* ln2b  = (const float*)d_in[22];
  const float* fcw   = (const float*)d_in[23];

  unsigned short* wsb = (unsigned short*)d_ws;
  float* he_g  = (float*)((char*)d_ws + WS_BYTES_BF);
  float* he2_g = he_g + 65536;

  auto cv = [&](const float* src, int off, int N, int Kr, int Kp){
    int total = N * Kp;
    prep_convert<<<(total + 255)/256, 256, 0, stream>>>(src, wsb + off, N, Kr, Kp);
  };
  cv(et1_w, O_ET1,          256, 256, 256);
  cv(wi1,   O_W1 + 0*W1SZ,  256, 528, 544);
  cv(wf1,   O_W1 + 1*W1SZ,  256, 528, 544);
  cv(wc1,   O_W1 + 2*W1SZ,  256, 528, 544);
  cv(wo1,   O_W1 + 3*W1SZ,  256, 528, 544);
  cv(et2_w, O_ET2,          256, 256, 256);
  cv(wi2,   O_W2 + 0*W2SZ,  256, 768, 768);
  cv(wf2,   O_W2 + 1*W2SZ,  256, 768, 768);
  cv(wc2,   O_W2 + 2*W2SZ,  256, 768, 768);
  cv(wo2,   O_W2 + 3*W2SZ,  256, 768, 768);
  prep_zero<<<(131072 + 255)/256, 256, 0, stream>>>(he_g, 131072);

  lstm2_main<<<16, 256, 0, stream>>>(
      x, wsb, he_g, he2_g,
      bi1, bf1, bc1, bo1, ln1w, ln1b,
      bi2, bf2, bc2, bo2, ln2w, ln2b,
      fcw, (float*)d_out);
}
